// NeuralNet_49452253446553
// MI455X (gfx1250) — matrix operand order repaired and checked by reading, not compiled, not run on hardware
//
#include <hip/hip_runtime.h>
#include <hip/hip_fp16.h>

typedef __attribute__((ext_vector_type(16))) _Float16 v16h;
typedef __attribute__((ext_vector_type(8)))  _Float16 v8h;
typedef __attribute__((ext_vector_type(8)))  float    v8f;

// ============================================================================
// Weight convert: W[K][64] f32 (row-major, as used by x @ W) -> Wh[64][K] f16
// (i.e. transposed so each output column's K entries are contiguous — exactly
// the 16-contiguous-half chunks a lane needs for a WMMA B fragment).
// ============================================================================
struct WConvArgs {
  const float* src[13];
  _Float16*    dst[13];
  int          K[13];
};

__global__ void k_wconv(WConvArgs a) {
  int m = blockIdx.x;
  const float* s = a.src[m];
  _Float16*    d = a.dst[m];
  int K   = a.K[m];
  int tot = K * 64;
  for (int i = threadIdx.x; i < tot; i += blockDim.x) {
    int k = i >> 6, n = i & 63;
    d[n * K + k] = (_Float16)s[i];
  }
}

// ============================================================================
// First embed layer (K = 5 or 19): prenorm + Linear + ReLU, plain f32 VALU.
// One thread per (row, out-col). Output f16 row-major [rows][64].
// ============================================================================
__global__ void k_embed1(const float* __restrict__ X, int nin,
                         const float* __restrict__ shift,
                         const float* __restrict__ scale,
                         const float* __restrict__ W1,
                         const float* __restrict__ b1,
                         _Float16* __restrict__ Y, int rows) {
  int t = blockIdx.x * blockDim.x + threadIdx.x;
  if (t >= rows * 64) return;
  int row = t >> 6, col = t & 63;
  float acc = b1[col];
  const float* xr = X + (size_t)row * nin;
  for (int k = 0; k < nin; ++k)
    acc += (xr[k] - shift[k]) * scale[k] * W1[k * 64 + col];
  Y[t] = (_Float16)fmaxf(acc, 0.f);
}

// ============================================================================
// Generic 16-row-tile linear: Y[rows,64] = act(X[rows,64] @ W + b)
// X f16 row-major, W pre-transposed f16 [64][64], Y f16.
// One wave per 16-row tile; 8 waves per 256-thread block.
// ============================================================================
__global__ void __launch_bounds__(256)
k_lin64(const _Float16* __restrict__ X, const _Float16* __restrict__ Wh,
        const float* __restrict__ bias, _Float16* __restrict__ Y,
        int relu, int ntiles) {
  int tile = blockIdx.x * 8 + (threadIdx.x >> 5);
  if (tile >= ntiles) return;           // wave-uniform
  int lane = threadIdx.x & 31;
  int hh   = lane >> 4;                 // half-wave select
  int lrow = lane & 15;
  const _Float16* xrow = X + (size_t)(tile * 16 + lrow) * 64;

  v8f acc[4] = {};
  for (int ks = 0; ks < 2; ++ks) {
    // A fragment (16x32 f16): lane holds row lrow, k = 16*hh + s (contiguous, same order as B)
    int c1 = ks * 32 + hh * 16;
    v8h lo = *(const v8h*)(xrow + c1);
    v8h hi = *(const v8h*)(xrow + c1 + 8);
    v16h a;
#pragma unroll
    for (int i = 0; i < 8; ++i) { a[i] = lo[i]; a[i + 8] = hi[i]; }
    int koff = ks * 32 + hh * 16;
#pragma unroll
    for (int nt = 0; nt < 4; ++nt) {
      v16h b = *(const v16h*)(Wh + (size_t)(nt * 16 + lrow) * 64 + koff);
      acc[nt] = __builtin_amdgcn_wmma_f32_16x16x32_f16(
          false, a, false, b, (short)0, acc[nt], false, false);
    }
  }
#pragma unroll
  for (int nt = 0; nt < 4; ++nt) {
    int col  = nt * 16 + lrow;
    float bv = bias ? bias[col] : 0.f;
#pragma unroll
    for (int r = 0; r < 8; ++r) {
      int row = tile * 16 + hh * 8 + r;
      float y = acc[nt][r] + bv;
      if (relu) y = fmaxf(y, 0.f);
      Y[(size_t)row * 64 + col] = (_Float16)y;
    }
  }
}

// ============================================================================
// Concat linear (K=128): Y = relu(concat(agg*sp, T) @ W + b), agg f32, T f16.
// W pre-transposed f16 [64][128].
// ============================================================================
__global__ void __launch_bounds__(256)
k_lincat(const float* __restrict__ A0, const float* __restrict__ sp_ptr,
         const _Float16* __restrict__ T, const _Float16* __restrict__ Wh,
         const float* __restrict__ bias, _Float16* __restrict__ Y, int ntiles) {
  int tile = blockIdx.x * 8 + (threadIdx.x >> 5);
  if (tile >= ntiles) return;
  int lane = threadIdx.x & 31;
  int hh   = lane >> 4;
  int lrow = lane & 15;
  float sp = *sp_ptr;
  const float*    arow = A0 + (size_t)(tile * 16 + lrow) * 64;
  const _Float16* trow = T  + (size_t)(tile * 16 + lrow) * 64;

  v8f acc[4] = {};
  for (int ks = 0; ks < 4; ++ks) {
    v16h a;
    if (ks < 2) {                       // K 0..63 from agg (f32, scaled by sp)
      int c1 = ks * 32 + hh * 16;
#pragma unroll
      for (int i = 0; i < 8; ++i) {
        a[i]     = (_Float16)(arow[c1 + i]      * sp);
        a[i + 8] = (_Float16)(arow[c1 + 8 + i] * sp);
      }
    } else {                            // K 64..127 from target feats (f16)
      int c1 = (ks - 2) * 32 + hh * 16;
      v8h lo = *(const v8h*)(trow + c1);
      v8h hi = *(const v8h*)(trow + c1 + 8);
#pragma unroll
      for (int i = 0; i < 8; ++i) { a[i] = lo[i]; a[i + 8] = hi[i]; }
    }
    int koff = ks * 32 + hh * 16;
#pragma unroll
    for (int nt = 0; nt < 4; ++nt) {
      v16h b = *(const v16h*)(Wh + (size_t)(nt * 16 + lrow) * 128 + koff);
      acc[nt] = __builtin_amdgcn_wmma_f32_16x16x32_f16(
          false, a, false, b, (short)0, acc[nt], false, false);
    }
  }
#pragma unroll
  for (int nt = 0; nt < 4; ++nt) {
    int col  = nt * 16 + lrow;
    float bv = bias[col];
#pragma unroll
    for (int r = 0; r < 8; ++r) {
      int row = tile * 16 + hh * 8 + r;
      Y[(size_t)row * 64 + col] = (_Float16)fmaxf(acc[nt][r] + bv, 0.f);
    }
  }
}

// ============================================================================
// Edge kernel: per 16-edge tile (one wave):
//   joint = L[ei0] + R[ei1] + e_norm * We        (gather, f32 accumulate)
//   msg   = relu(joint * sf) @ Wf + bf           (WMMA f16->f32)
//   agg[tgt] += msg                              (global f32 atomics)
// ============================================================================
__global__ void __launch_bounds__(256)
k_edge(const _Float16* __restrict__ L, const _Float16* __restrict__ R,
       const float* __restrict__ efeat, const int* __restrict__ ei0,
       const int* __restrict__ ei1, const int* __restrict__ eit,
       const float* __restrict__ eshift, const float* __restrict__ escale,
       const float* __restrict__ We, const float* __restrict__ sf_ptr,
       const _Float16* __restrict__ Wf, const float* __restrict__ bf,
       float* __restrict__ agg, int ntiles) {
  int tile = blockIdx.x * 8 + (threadIdx.x >> 5);
  if (tile >= ntiles) return;
  int lane = threadIdx.x & 31;
  int hh   = lane >> 4;
  int lrow = lane & 15;

  int er   = tile * 16 + lrow;          // edge whose A-row this lane builds
  int i0   = ei0[er];
  int i1   = ei1[er];
  float ef = (efeat[er] - eshift[0]) * escale[0];
  float sf = *sf_ptr;
  const _Float16* lr = L + (size_t)i0 * 64;
  const _Float16* rr = R + (size_t)i1 * 64;

  v8f acc[4] = {};
  for (int ks = 0; ks < 2; ++ks) {
    int c1 = ks * 32 + hh * 16;
    int c2 = c1 + 8;
    v8h l1 = *(const v8h*)(lr + c1);
    v8h l2 = *(const v8h*)(lr + c2);
    v8h r1 = *(const v8h*)(rr + c1);
    v8h r2 = *(const v8h*)(rr + c2);
    v16h a;
#pragma unroll
    for (int i = 0; i < 8; ++i) {
      float j1 = ((float)l1[i] + (float)r1[i] + ef * We[c1 + i]) * sf;
      float j2 = ((float)l2[i] + (float)r2[i] + ef * We[c2 + i]) * sf;
      a[i]     = (_Float16)fmaxf(j1, 0.f);
      a[i + 8] = (_Float16)fmaxf(j2, 0.f);
    }
    int koff = ks * 32 + hh * 16;
#pragma unroll
    for (int nt = 0; nt < 4; ++nt) {
      v16h b = *(const v16h*)(Wf + (size_t)(nt * 16 + lrow) * 64 + koff);
      acc[nt] = __builtin_amdgcn_wmma_f32_16x16x32_f16(
          false, a, false, b, (short)0, acc[nt], false, false);
    }
  }

  int tgt[8];
#pragma unroll
  for (int r = 0; r < 8; ++r) tgt[r] = eit[tile * 16 + hh * 8 + r];
#pragma unroll
  for (int nt = 0; nt < 4; ++nt) {
    int col  = nt * 16 + lrow;
    float bv = bf[col];
#pragma unroll
    for (int r = 0; r < 8; ++r)
      unsafeAtomicAdd(&agg[(size_t)tgt[r] * 64 + col], acc[nt][r] + bv);
  }
}

// ============================================================================
// Output head: out[row] = Y1[row,:] . W2   (Y1 already relu'd), f32 result.
// ============================================================================
__global__ void k_out(const _Float16* __restrict__ Y1,
                      const float* __restrict__ W2,
                      float* __restrict__ out, int rows) {
  int row = blockIdx.x * blockDim.x + threadIdx.x;
  if (row >= rows) return;
  const _Float16* yr = Y1 + (size_t)row * 64;
  float acc = 0.f;
#pragma unroll 8
  for (int k = 0; k < 64; ++k) acc += (float)yr[k] * W2[k];
  out[row] = acc;
}

// ============================================================================
extern "C" void kernel_launch(void* const* d_in, const int* in_sizes, int n_in,
                              void* d_out, int out_size, void* d_ws, size_t ws_size,
                              hipStream_t stream) {
  (void)n_in; (void)out_size; (void)ws_size;

  // ---- inputs, setup_inputs() dict order ----
  const float* cons   = (const float*)d_in[0];
  const float* efeat  = (const float*)d_in[1];
  const float* vars   = (const float*)d_in[2];
  const float* c_sh   = (const float*)d_in[3];
  const float* c_sc   = (const float*)d_in[4];
  const float* v_sh   = (const float*)d_in[5];
  const float* v_sc   = (const float*)d_in[6];
  const float* e_sh   = (const float*)d_in[7];
  const float* e_sc   = (const float*)d_in[8];
  const float* c_W1   = (const float*)d_in[9];
  const float* c_b1   = (const float*)d_in[10];
  const float* c_b2   = (const float*)d_in[12];
  const float* v_W1   = (const float*)d_in[13];
  const float* v_b1   = (const float*)d_in[14];
  const float* v_b2   = (const float*)d_in[16];
  const float* vc_bl  = (const float*)d_in[18];
  const float* vc_We  = (const float*)d_in[19];
  const float* vc_sf  = (const float*)d_in[21];
  const float* vc_bf  = (const float*)d_in[23];
  const float* vc_sp  = (const float*)d_in[24];
  const float* vc_boa = (const float*)d_in[26];
  const float* vc_bob = (const float*)d_in[28];
  const float* cv_bl  = (const float*)d_in[30];
  const float* cv_We  = (const float*)d_in[31];
  const float* cv_sf  = (const float*)d_in[33];
  const float* cv_bf  = (const float*)d_in[35];
  const float* cv_sp  = (const float*)d_in[36];
  const float* cv_boa = (const float*)d_in[38];
  const float* cv_bob = (const float*)d_in[40];
  const float* out_b1 = (const float*)d_in[42];
  const float* out_W2 = (const float*)d_in[43];
  const int*   eidx   = (const int*)  d_in[44];

  const int n_cons  = in_sizes[0] / 5;
  const int n_vars  = in_sizes[2] / 19;
  const int n_edges = in_sizes[1];
  const int* ei0 = eidx;
  const int* ei1 = eidx + n_edges;

  // ---- workspace carving (all 256B aligned) ----
  size_t off = 0;
  char* base = (char*)d_ws;
  auto carve = [&](size_t bytes) -> void* {
    void* p = base + off;
    off += (bytes + 255) & ~(size_t)255;
    return p;
  };
  _Float16* WH   = (_Float16*)carve((size_t)13 * 8192 * sizeof(_Float16));
  _Float16* cEmb = (_Float16*)carve((size_t)n_cons * 64 * 2);
  _Float16* vEmb = (_Float16*)carve((size_t)n_vars * 64 * 2);
  _Float16* Lbuf = (_Float16*)carve((size_t)n_cons * 64 * 2);
  _Float16* Rbuf = (_Float16*)carve((size_t)n_vars * 64 * 2);
  _Float16* c2   = (_Float16*)carve((size_t)n_cons * 64 * 2);
  _Float16* v2   = (_Float16*)carve((size_t)n_vars * 64 * 2);
  _Float16* tmp  = (_Float16*)carve((size_t)n_vars * 64 * 2);
  float*    agg  = (float*)   carve((size_t)n_vars * 64 * 4);

  // ---- convert 13 weight matrices to transposed f16 ----
  WConvArgs wa;
  const int widx[13] = {11, 15, 17, 20, 22, 25, 27, 29, 32, 34, 37, 39, 41};
  const int wK[13]   = {64, 64, 64, 64, 64, 128, 64, 64, 64, 64, 128, 64, 64};
  for (int m = 0; m < 13; ++m) {
    wa.src[m] = (const float*)d_in[widx[m]];
    wa.dst[m] = WH + (size_t)m * 8192;
    wa.K[m]   = wK[m];
  }
  // slots: 0=cW2 1=vW2 2=vcWl 3=vcWr 4=vcWf 5=vcWoa 6=vcWob
  //        7=cvWl 8=cvWr 9=cvWf 10=cvWoa 11=cvWob 12=outW1
  k_wconv<<<13, 256, 0, stream>>>(wa);

  const int ct = n_cons / 16, vt = n_vars / 16, et = n_edges / 16;
  auto gb = [](int tiles) { return (tiles + 7) / 8; };

  // ---- embeds ----
  k_embed1<<<(n_cons * 64 + 255) / 256, 256, 0, stream>>>(
      cons, 5, c_sh, c_sc, c_W1, c_b1, tmp, n_cons);
  k_lin64<<<gb(ct), 256, 0, stream>>>(tmp, WH + 0 * 8192, c_b2, cEmb, 1, ct);
  k_embed1<<<(n_vars * 64 + 255) / 256, 256, 0, stream>>>(
      vars, 19, v_sh, v_sc, v_W1, v_b1, tmp, n_vars);
  k_lin64<<<gb(vt), 256, 0, stream>>>(tmp, WH + 1 * 8192, v_b2, vEmb, 1, vt);

  // ---- conv v->c (target = constraints via ei0) ----
  k_lin64<<<gb(ct), 256, 0, stream>>>(cEmb, WH + 2 * 8192, vc_bl, Lbuf, 0, ct);
  k_lin64<<<gb(vt), 256, 0, stream>>>(vEmb, WH + 3 * 8192, nullptr, Rbuf, 0, vt);
  hipMemsetAsync(agg, 0, (size_t)n_cons * 64 * 4, stream);
  k_edge<<<gb(et), 256, 0, stream>>>(Lbuf, Rbuf, efeat, ei0, ei1, ei0,
                                     e_sh, e_sc, vc_We, vc_sf,
                                     WH + 4 * 8192, vc_bf, agg, et);
  k_lincat<<<gb(ct), 256, 0, stream>>>(agg, vc_sp, cEmb, WH + 5 * 8192,
                                       vc_boa, tmp, ct);
  k_lin64<<<gb(ct), 256, 0, stream>>>(tmp, WH + 6 * 8192, vc_bob, c2, 1, ct);

  // ---- conv c->v (left = new c, target = variables via ei1) ----
  k_lin64<<<gb(ct), 256, 0, stream>>>(c2, WH + 7 * 8192, cv_bl, Lbuf, 0, ct);
  k_lin64<<<gb(vt), 256, 0, stream>>>(vEmb, WH + 8 * 8192, nullptr, Rbuf, 0, vt);
  hipMemsetAsync(agg, 0, (size_t)n_vars * 64 * 4, stream);
  k_edge<<<gb(et), 256, 0, stream>>>(Lbuf, Rbuf, efeat, ei0, ei1, ei1,
                                     e_sh, e_sc, cv_We, cv_sf,
                                     WH + 9 * 8192, cv_bf, agg, et);
  k_lincat<<<gb(vt), 256, 0, stream>>>(agg, cv_sp, vEmb, WH + 10 * 8192,
                                       cv_boa, tmp, vt);
  k_lin64<<<gb(vt), 256, 0, stream>>>(tmp, WH + 11 * 8192, cv_bob, v2, 1, vt);

  // ---- output head ----
  k_lin64<<<gb(vt), 256, 0, stream>>>(v2, WH + 12 * 8192, out_b1, Rbuf, 1, vt);
  k_out<<<(n_vars + 255) / 256, 256, 0, stream>>>(Rbuf, out_W2,
                                                  (float*)d_out, n_vars);
}
